// VisionMambaBackbone_83537113907380
// MI455X (gfx1250) — matrix operand order repaired and checked by reading, not compiled, not run on hardware
//
#include <hip/hip_runtime.h>
#include <hip/hip_bf16.h>
#include <math.h>

#define B_SZ 4
#define L_TOK 1024
#define NTOK (B_SZ * L_TOK)   // 4096 tokens
#define DM 256                // d_model
#define DI 512                // d_inner
#define DSTATE 16
#define XDBL 64               // dt_rank(32) + 2*16
#define NH 8
#define HD 32

typedef __attribute__((ext_vector_type(16))) _Float16 v16h;
typedef __attribute__((ext_vector_type(8)))  float    v8f;
typedef __attribute__((ext_vector_type(2)))  _Float16 h2f;

union H16 { v16h v; _Float16 h[16]; unsigned u[8]; };
union H2U { h2f v; unsigned u; };

__device__ __forceinline__ float dev_silu(float x) { return x / (1.f + __expf(-x)); }
__device__ __forceinline__ float dev_softplus(float x) { return x > 20.f ? x : log1pf(__expf(x)); }

// packed f16x2 xor-shuffle max (one ds op + packed f16 max per step, 2 rows at once)
__device__ __forceinline__ h2f h2max_xor(h2f v, int m) {
    H2U a; a.v = v;
    H2U b; b.u = __shfl_xor(a.u, m, 32);
    h2f r;
    r.x = (v.x > b.v.x) ? v.x : b.v.x;
    r.y = (v.y > b.v.y) ? v.y : b.v.y;
    return r;
}

// ---------------------------------------------------------------------------
// WMMA GEMM:  C[M,N] = act( A[M,K](lda) @ W[K,N] + bias ) (+R)
// 256 threads = 8 waves (4x2 grid of waves). Workgroup tile 128(M) x 64(N);
// each wave computes a 32x32 tile = 4 independent v_wmma_f32_16x16x32_f16
// per 32-wide K step (2 A-frags x 2 B-frags).
// ---------------------------------------------------------------------------
template<int ACT, bool BIAS, bool RES>
__global__ __launch_bounds__(256) void gemm_wmma(
    const float* __restrict__ A, const float* __restrict__ W,
    const float* __restrict__ bias, const float* R,
    float* C, int K, int lda, int N, int ldc)
{
    __shared__ _Float16 lA[128 * 32];  // [row][k]
    __shared__ _Float16 lBt[64 * 32];  // [n][k]  (transposed)

    const int tid  = threadIdx.x;
    const int m0   = blockIdx.y * 128;
    const int n0   = blockIdx.x * 64;
    const int wv   = tid >> 5;
    const int lane = tid & 31;
    const int half = lane >> 4;
    const int ml   = lane & 15;
    const int wm   = (wv & 3) * 32;   // wave M offset in tile
    const int wn   = (wv >> 2) * 32;  // wave N offset in tile

    v8f acc00 = {0.f,0.f,0.f,0.f,0.f,0.f,0.f,0.f};
    v8f acc01 = acc00, acc10 = acc00, acc11 = acc00;

    const int ar = tid >> 1;          // 0..127 A tile row
    const int ac = (tid & 1) * 16;    // 0/16   A tile col group (16 floats)
    const int br = tid >> 3;          // 0..31  B tile row (k)
    const int bc = (tid & 7) * 8;     // 0..56  B tile col group (8 floats)

    for (int k0 = 0; k0 < K; k0 += 32) {
        {
            const float* ap = A + (size_t)(m0 + ar) * lda + k0 + ac;
            const float4 a0 = *(const float4*)(ap + 0);
            const float4 a1 = *(const float4*)(ap + 4);
            const float4 a2 = *(const float4*)(ap + 8);
            const float4 a3 = *(const float4*)(ap + 12);
            _Float16* pa = &lA[ar * 32 + ac];
            pa[0]  = (_Float16)a0.x; pa[1]  = (_Float16)a0.y;
            pa[2]  = (_Float16)a0.z; pa[3]  = (_Float16)a0.w;
            pa[4]  = (_Float16)a1.x; pa[5]  = (_Float16)a1.y;
            pa[6]  = (_Float16)a1.z; pa[7]  = (_Float16)a1.w;
            pa[8]  = (_Float16)a2.x; pa[9]  = (_Float16)a2.y;
            pa[10] = (_Float16)a2.z; pa[11] = (_Float16)a2.w;
            pa[12] = (_Float16)a3.x; pa[13] = (_Float16)a3.y;
            pa[14] = (_Float16)a3.z; pa[15] = (_Float16)a3.w;

            const float* wp = W + (size_t)(k0 + br) * N + n0 + bc;
            const float4 b0 = *(const float4*)(wp + 0);
            const float4 b1 = *(const float4*)(wp + 4);
            lBt[(bc + 0) * 32 + br] = (_Float16)b0.x;
            lBt[(bc + 1) * 32 + br] = (_Float16)b0.y;
            lBt[(bc + 2) * 32 + br] = (_Float16)b0.z;
            lBt[(bc + 3) * 32 + br] = (_Float16)b0.w;
            lBt[(bc + 4) * 32 + br] = (_Float16)b1.x;
            lBt[(bc + 5) * 32 + br] = (_Float16)b1.y;
            lBt[(bc + 6) * 32 + br] = (_Float16)b1.z;
            lBt[(bc + 7) * 32 + br] = (_Float16)b1.w;
        }
        __syncthreads();

        H16 a0f, a1f, b0f, b1f;
#pragma unroll
        for (int i = 0; i < 8; ++i) {
            const int kk = half * 16 + 2 * i;
            a0f.u[i] = *(const unsigned*)&lA[(wm + ml) * 32 + kk];
            a1f.u[i] = *(const unsigned*)&lA[(wm + 16 + ml) * 32 + kk];
            b0f.u[i] = *(const unsigned*)&lBt[(wn + ml) * 32 + half * 16 + 2 * i];
            b1f.u[i] = *(const unsigned*)&lBt[(wn + 16 + ml) * 32 + half * 16 + 2 * i];
        }
        acc00 = __builtin_amdgcn_wmma_f32_16x16x32_f16(false, a0f.v, false, b0f.v,
                                                       (short)0, acc00, false, false);
        acc01 = __builtin_amdgcn_wmma_f32_16x16x32_f16(false, a0f.v, false, b1f.v,
                                                       (short)0, acc01, false, false);
        acc10 = __builtin_amdgcn_wmma_f32_16x16x32_f16(false, a1f.v, false, b0f.v,
                                                       (short)0, acc10, false, false);
        acc11 = __builtin_amdgcn_wmma_f32_16x16x32_f16(false, a1f.v, false, b1f.v,
                                                       (short)0, acc11, false, false);
        __syncthreads();
    }

    auto emit = [&](v8f acc, int rbase, int cbase) {
        const int col = cbase + ml;
#pragma unroll
        for (int j = 0; j < 8; ++j) {
            const int row = rbase + half * 8 + j;
            float v = acc[j];
            if (BIAS) v += bias[col];
            if (ACT == 1) v = dev_silu(v);
            if (ACT == 2) v = dev_softplus(v);
            if (RES) v += R[(size_t)row * ldc + col];
            C[(size_t)row * ldc + col] = v;
        }
    };
    emit(acc00, m0 + wm,      n0 + wn);
    emit(acc01, m0 + wm,      n0 + wn + 16);
    emit(acc10, m0 + wm + 16, n0 + wn);
    emit(acc11, m0 + wm + 16, n0 + wn + 16);
}

// ---------------------------------------------------------------------------
// Flash cross-attention with WMMA.  hd = 32 == WMMA K.  One wave owns 16 q
// rows; the block (8 waves) shares one (b,h): K/V chunks staged through LDS.
// Row sums of P computed with a WMMA against an all-ones B fragment (replaces
// 32 ds_bpermutes/chunk); row max reduced as packed f16x2, exact for softmax
// since the same shift is applied lane-uniformly per row.
// ---------------------------------------------------------------------------
__global__ __launch_bounds__(256) void attn_wmma(
    const float* __restrict__ Q, const float* __restrict__ Km,
    const float* __restrict__ V, float* __restrict__ O)
{
    __shared__ _Float16 lK[32 * 32];   // [krow][hd]
    __shared__ _Float16 lVt[32 * 32];  // [hd][krow]
    __shared__ _Float16 lp[8 * 16 * 32];

    const int tid  = threadIdx.x;
    const int wv   = tid >> 5;
    const int lane = tid & 31;
    const int half = lane >> 4;
    const int ml   = lane & 15;
    const int blk  = blockIdx.x;
    const int qt = blk & 7, h = (blk >> 3) & 7, b = blk >> 6;
    const int q0 = qt * 128 + wv * 16;
    const float scale = 0.1767766953f;  // 1/sqrt(32)

    H16 qa, ones;
#pragma unroll
    for (int e = 0; e < 16; ++e) ones.h[e] = (_Float16)1.f;
    {
        const float* qp = Q + ((size_t)(b * L_TOK + q0 + ml)) * DM + h * HD;
#pragma unroll
        for (int i = 0; i < 8; ++i) {
            const int kk = half * 16 + 2 * i;
            const float2 f = *(const float2*)(qp + kk);
            qa.h[2 * i]     = (_Float16)(f.x * scale);
            qa.h[2 * i + 1] = (_Float16)(f.y * scale);
        }
    }

    float M8[8], L8[8];
#pragma unroll
    for (int j = 0; j < 8; ++j) { M8[j] = -1e30f; L8[j] = 0.f; }
    v8f o0 = {0.f,0.f,0.f,0.f,0.f,0.f,0.f,0.f};
    v8f o1 = o0;
    _Float16* myp = &lp[wv * 512];

    const int cr = tid >> 3;        // 0..31 cooperative row
    const int cc = (tid & 7) * 4;   // 0..28 cooperative col group

    for (int kc = 0; kc < L_TOK; kc += 32) {
        {
            const float4 kf = *(const float4*)(Km + (size_t)(b * L_TOK + kc + cr) * DM + h * HD + cc);
            lK[cr * 32 + cc + 0] = (_Float16)kf.x;
            lK[cr * 32 + cc + 1] = (_Float16)kf.y;
            lK[cr * 32 + cc + 2] = (_Float16)kf.z;
            lK[cr * 32 + cc + 3] = (_Float16)kf.w;
            const float4 vf = *(const float4*)(V + (size_t)(b * L_TOK + kc + cr) * DM + h * HD + cc);
            lVt[(cc + 0) * 32 + cr] = (_Float16)vf.x;
            lVt[(cc + 1) * 32 + cr] = (_Float16)vf.y;
            lVt[(cc + 2) * 32 + cr] = (_Float16)vf.z;
            lVt[(cc + 3) * 32 + cr] = (_Float16)vf.w;
        }
        __syncthreads();

        H16 kb0, kb1, vb0, vb1;
#pragma unroll
        for (int i = 0; i < 8; ++i) {
            const int o = half * 16 + 2 * i;
            kb0.u[i] = *(const unsigned*)&lK[ml * 32 + o];
            kb1.u[i] = *(const unsigned*)&lK[(16 + ml) * 32 + o];
            vb0.u[i] = *(const unsigned*)&lVt[ml * 32 + o];
            vb1.u[i] = *(const unsigned*)&lVt[(16 + ml) * 32 + o];
        }
        const v8f z8 = {0.f,0.f,0.f,0.f,0.f,0.f,0.f,0.f};
        v8f s0 = __builtin_amdgcn_wmma_f32_16x16x32_f16(false, qa.v, false, kb0.v,
                                                        (short)0, z8, false, false);
        v8f s1 = __builtin_amdgcn_wmma_f32_16x16x32_f16(false, qa.v, false, kb1.v,
                                                        (short)0, z8, false, false);

        // row max: packed f16x2 xor-reduction over the 16 lanes of each half
        float cor8[8];
#pragma unroll
        for (int jp = 0; jp < 4; ++jp) {
            const int j0 = 2 * jp, j1 = 2 * jp + 1;
            h2f hm;
            hm.x = (_Float16)fmaxf(s0[j0], s1[j0]);
            hm.y = (_Float16)fmaxf(s0[j1], s1[j1]);
            hm = h2max_xor(hm, 1);
            hm = h2max_xor(hm, 2);
            hm = h2max_xor(hm, 4);
            hm = h2max_xor(hm, 8);
            const float mnew0 = fmaxf(M8[j0], (float)hm.x);
            const float mnew1 = fmaxf(M8[j1], (float)hm.y);
            cor8[j0] = __expf(M8[j0] - mnew0); M8[j0] = mnew0;
            cor8[j1] = __expf(M8[j1] - mnew1); M8[j1] = mnew1;
        }
#pragma unroll
        for (int j = 0; j < 8; ++j) {
            const float p0 = __expf(s0[j] - M8[j]);
            const float p1 = __expf(s1[j] - M8[j]);
            o0[j] = o0[j] * cor8[j];
            o1[j] = o1[j] * cor8[j];
            myp[(j + 8 * half) * 32 + ml]      = (_Float16)p0;
            myp[(j + 8 * half) * 32 + 16 + ml] = (_Float16)p1;
        }
        asm volatile("s_wait_dscnt 0" ::: "memory");
        H16 pa;
#pragma unroll
        for (int i = 0; i < 8; ++i) {
            const int kk = half * 16 + 2 * i;
            pa.u[i] = *(const unsigned*)&myp[ml * 32 + kk];
        }
        // row sums of P via WMMA with all-ones B (every output column == rowsum)
        v8f sums = __builtin_amdgcn_wmma_f32_16x16x32_f16(false, pa.v, false, ones.v,
                                                          (short)0, z8, false, false);
#pragma unroll
        for (int j = 0; j < 8; ++j) L8[j] = L8[j] * cor8[j] + sums[j];

        o0 = __builtin_amdgcn_wmma_f32_16x16x32_f16(false, pa.v, false, vb0.v,
                                                    (short)0, o0, false, false);
        o1 = __builtin_amdgcn_wmma_f32_16x16x32_f16(false, pa.v, false, vb1.v,
                                                    (short)0, o1, false, false);
        __syncthreads();
    }
#pragma unroll
    for (int j = 0; j < 8; ++j) {
        const float inv = 1.f / L8[j];
        float* op = O + ((size_t)(b * L_TOK + q0 + half * 8 + j)) * DM + h * HD;
        op[ml]      = o0[j] * inv;
        op[16 + ml] = o1[j] * inv;
    }
}

// ---------------------------------------------------------------------------
// Selective scan, both directions. grid = B*2, 512 threads = one channel each.
// 16-state recurrence in VGPRs; B/C loads are wave-uniform (scalarizable).
// ---------------------------------------------------------------------------
__global__ __launch_bounds__(512) void scan_kernel(
    const float* __restrict__ xact,
    const float* __restrict__ xdbl_f, const float* __restrict__ dt_f,
    const float* __restrict__ Alog_f, const float* __restrict__ D_f,
    const float* __restrict__ xdbl_b, const float* __restrict__ dt_b,
    const float* __restrict__ Alog_b, const float* __restrict__ D_b,
    float* __restrict__ y_f, float* __restrict__ y_b)
{
    const int d   = threadIdx.x;
    const int b   = blockIdx.x >> 1;
    const int dir = blockIdx.x & 1;
    const float* xd = dir ? xdbl_b : xdbl_f;
    const float* dt = dir ? dt_b : dt_f;
    const float* Al = dir ? Alog_b : Alog_f;
    const float* Dp = dir ? D_b : D_f;
    float* yo = dir ? y_b : y_f;

    float A[16], hst[16];
#pragma unroll
    for (int n = 0; n < 16; ++n) { A[n] = -__expf(Al[d * 16 + n]); hst[n] = 0.f; }
    const float Dd = Dp[d];

    for (int s = 0; s < L_TOK; ++s) {
        const int l = dir ? (L_TOK - 1 - s) : s;
        const size_t idx = (size_t)b * L_TOK + l;
        const float dtv = dt[idx * DI + d];
        const float xv  = xact[idx * DI + d];
        const float* bcv = xd + idx * XDBL;
        float y = 0.f;
#pragma unroll
        for (int n = 0; n < 16; ++n) {
            const float dA = __expf(dtv * A[n]);
            hst[n] = dA * hst[n] + dtv * bcv[32 + n] * xv;
            y += hst[n] * bcv[48 + n];
        }
        yo[idx * DI + d] = y + xv * Dd;
    }
}

// ---------------------------------------------------------------------------
// LayerNorm over D=256. One wave per token, 8 waves per block. In-place safe.
// ---------------------------------------------------------------------------
__global__ __launch_bounds__(256) void ln_kernel(
    const float* X, const float* __restrict__ g, const float* __restrict__ bta,
    float* Yo)
{
    const int wv = threadIdx.x >> 5, lane = threadIdx.x & 31;
    const int t = blockIdx.x * 8 + wv;
    const float* x = X + (size_t)t * DM;
    float v[8]; float s = 0.f, sq = 0.f;
#pragma unroll
    for (int i = 0; i < 8; ++i) { v[i] = x[lane + 32 * i]; s += v[i]; sq += v[i] * v[i]; }
#pragma unroll
    for (int m = 1; m <= 16; m <<= 1) { s += __shfl_xor(s, m, 32); sq += __shfl_xor(sq, m, 32); }
    const float mean = s * (1.f / 256.f);
    const float var = sq * (1.f / 256.f) - mean * mean;
    const float r = rsqrtf(var + 1e-5f);
    float* yo = Yo + (size_t)t * DM;
#pragma unroll
    for (int i = 0; i < 8; ++i)
        yo[lane + 32 * i] = (v[i] - mean) * r * g[lane + 32 * i] + bta[lane + 32 * i];
}

// --------------------------- small scalar kernels ---------------------------
__global__ void time_mlp1_kernel(const float* __restrict__ te, const float* __restrict__ W,
                                 const float* __restrict__ b, float* o) {
    const int gid = blockIdx.x * 256 + threadIdx.x;  // 0..4095
    const int bi = gid >> 10, oi = gid & 1023;
    float acc = b[oi];
    for (int k = 0; k < 256; ++k) acc += te[bi * 256 + k] * W[k * 1024 + oi];
    o[gid] = dev_silu(acc);
}

__global__ void time_mlp2_kernel(const float* __restrict__ t1, const float* __restrict__ W,
                                 const float* __restrict__ b, float* o) {
    const int gid = blockIdx.x * 256 + threadIdx.x;  // 0..1023
    const int bi = gid >> 8, oi = gid & 255;
    float acc = b[oi];
    for (int k = 0; k < 1024; ++k) acc += t1[bi * 1024 + k] * W[k * 256 + oi];
    o[gid] = acc;
}

__global__ void embed_kernel(const float* __restrict__ z, const float* __restrict__ cm,
                             const float* __restrict__ w, const float* __restrict__ bb,
                             const float* __restrict__ tv, float* tok) {
    const int t = blockIdx.x, d = threadIdx.x;
    const int b = t >> 10, l = t & 1023;
    float acc = bb[d] + tv[b * 256 + d];
#pragma unroll
    for (int c = 0; c < 13; ++c) acc += z[((b * 13 + c) << 10) + l] * w[c * 256 + d];
    acc += cm[(b << 10) + l] * w[13 * 256 + d];
    tok[(size_t)t * 256 + d] = acc;
}

__global__ void transpose_sar(const float* __restrict__ s, float* o) {
    const int gid = blockIdx.x * 256 + threadIdx.x;   // NTOK*256
    const int row = gid >> 8, c = gid & 255;          // row = b*1024+l
    const int b = row >> 10, l = row & 1023;
    o[gid] = s[((size_t)(b * 256 + c)) * 1024 + l];
}

__global__ void conv_kernel(const float* __restrict__ xz, const float* __restrict__ w,
                            const float* __restrict__ bconv, float* xa) {
    const int gid = blockIdx.x * 256 + threadIdx.x;   // B*L*512
    const int c = gid & 511, l = (gid >> 9) & 1023, b = gid >> 19;
    const size_t rb = (size_t)b * L_TOK;
    float acc = bconv[c] + w[c * 3 + 1] * xz[(rb + l) * 1024 + c];
    if (l > 0)    acc += w[c * 3 + 0] * xz[(rb + l - 1) * 1024 + c];
    if (l < 1023) acc += w[c * 3 + 2] * xz[(rb + l + 1) * 1024 + c];
    xa[(rb + l) * DI + c] = dev_silu(acc);
}

__global__ void gate_kernel(const float* __restrict__ yf, const float* __restrict__ yb,
                            const float* __restrict__ xz, float* yo) {
    const int gid = blockIdx.x * 256 + threadIdx.x;  // NTOK*512
    const int c = gid & 511, row = gid >> 9;
    const float z = xz[((size_t)row << 10) + 512 + c];
    yo[gid] = (yf[gid] + yb[gid]) * dev_silu(z);
}

__global__ void unembed_kernel(const float* __restrict__ tok, const float* __restrict__ w,
                               const float* __restrict__ bb, float* out) {
    const int gid = blockIdx.x * 256 + threadIdx.x;
    if (gid >= NTOK * 13) return;
    const int c = gid % 13, t = gid / 13;
    float acc = bb[c];
    for (int d = 0; d < 256; ++d) acc += tok[(size_t)t * 256 + d] * w[d * 13 + c];
    const int b = t >> 10, l = t & 1023, hh = l >> 5, wi = l & 31;
    out[((b * 13 + c) * 32 + hh) * 32 + wi] = acc;
}

// ---------------------------------------------------------------------------
extern "C" void kernel_launch(void* const* d_in, const int* in_sizes, int n_in,
                              void* d_out, int out_size, void* d_ws, size_t ws_size,
                              hipStream_t stream) {
    auto F = [&](int i) { return (const float*)d_in[i]; };
    const float* z_t   = F(0);
    const float* sar   = F(1);
    const float* t_emb = F(2);
    const float* cloud = F(3);

    // jax pytree (sorted-key) flattening of params: blocks first
    int bb[12];
    { int p = 4; for (int i = 0; i < 12; ++i) { bb[i] = p; p += (i % 3 == 0) ? 26 : 16; } }
    // post-block leaves at index 236:
    // 236 norm_out.b 237 norm_out.g 238 pe.b 239 pe.w 240 un.b 241 un.w
    // 242 sar.b 243 sar.w 244 tm1.b 245 tm1.w 246 tm2.b 247 tm2.w

    float* ws = (float*)d_ws;
    float* t1      = ws;                          // 4096
    float* tvec    = t1 + 4096;                   // 1024
    float* tokens  = tvec + 1024;                 // NTOK*DM
    float* xnorm   = tokens + (size_t)NTOK * DM;
    float* sar_t   = xnorm + (size_t)NTOK * DM;
    float* sar_tok = sar_t + (size_t)NTOK * DM;
    float* xz      = sar_tok + (size_t)NTOK * DM; // NTOK*1024
    float* xact    = xz + (size_t)NTOK * 1024;    // NTOK*512
    float* xdf     = xact + (size_t)NTOK * DI;    // NTOK*64
    float* xdb     = xdf + (size_t)NTOK * XDBL;
    float* dtf     = xdb + (size_t)NTOK * XDBL;   // NTOK*512
    float* dtb     = dtf + (size_t)NTOK * DI;
    float* yf      = dtb + (size_t)NTOK * DI;
    float* yb      = yf + (size_t)NTOK * DI;
    float* ygate   = dtf;                         // reuse (dt dead after scan)
    float* qb = xz;                               // reuse (xz dead after gate)
    float* kb = xz + (size_t)NTOK * DM;
    float* vb = xz + 2 * (size_t)NTOK * DM;
    float* ab = xz + 3 * (size_t)NTOK * DM;

    // time MLP + embedding
    time_mlp1_kernel<<<16, 256, 0, stream>>>(t_emb, F(245), F(244), t1);
    time_mlp2_kernel<<<4, 256, 0, stream>>>(t1, F(247), F(246), tvec);
    embed_kernel<<<NTOK, 256, 0, stream>>>(z_t, cloud, F(239), F(238), tvec, tokens);

    // SAR tokens
    transpose_sar<<<NTOK * DM / 256, 256, 0, stream>>>(sar, sar_t);
    gemm_wmma<0, true, false><<<dim3(DM / 64, NTOK / 128), 256, 0, stream>>>(
        sar_t, F(243), F(242), nullptr, sar_tok, 256, 256, 256, 256);

    for (int i = 0; i < 12; ++i) {
        const int B0 = bb[i];
        // block leaves: 0 conv_b 1 conv_w 2 in_proj.w 3 norm.b 4 norm.g 5 out_proj.w
        // 6..10 bwd{A_log,D,dt.b,dt.w,x.w} 11..15 fwd{...}
        // xattn: 16 k.b 17 k.w 18 n.b 19 n.g 20 o.b 21 o.w 22 q.b 23 q.w 24 v.b 25 v.w
        ln_kernel<<<512, 256, 0, stream>>>(tokens, F(B0 + 4), F(B0 + 3), xnorm);
        gemm_wmma<0, false, false><<<dim3(1024 / 64, NTOK / 128), 256, 0, stream>>>(
            xnorm, F(B0 + 2), nullptr, nullptr, xz, 256, 256, 1024, 1024);
        conv_kernel<<<NTOK * DI / 256, 256, 0, stream>>>(xz, F(B0 + 1), F(B0 + 0), xact);
        gemm_wmma<0, false, false><<<dim3(XDBL / 64, NTOK / 128), 256, 0, stream>>>(
            xact, F(B0 + 15), nullptr, nullptr, xdf, 512, 512, 64, 64);
        gemm_wmma<0, false, false><<<dim3(XDBL / 64, NTOK / 128), 256, 0, stream>>>(
            xact, F(B0 + 10), nullptr, nullptr, xdb, 512, 512, 64, 64);
        gemm_wmma<2, true, false><<<dim3(DI / 64, NTOK / 128), 256, 0, stream>>>(
            xdf, F(B0 + 14), F(B0 + 13), nullptr, dtf, 32, 64, 512, 512);
        gemm_wmma<2, true, false><<<dim3(DI / 64, NTOK / 128), 256, 0, stream>>>(
            xdb, F(B0 + 9), F(B0 + 8), nullptr, dtb, 32, 64, 512, 512);
        scan_kernel<<<B_SZ * 2, 512, 0, stream>>>(
            xact, xdf, dtf, F(B0 + 11), F(B0 + 12),
            xdb, dtb, F(B0 + 6), F(B0 + 7), yf, yb);
        gate_kernel<<<NTOK * DI / 256, 256, 0, stream>>>(yf, yb, xz, ygate);
        gemm_wmma<0, false, true><<<dim3(DM / 64, NTOK / 128), 256, 0, stream>>>(
            ygate, F(B0 + 5), nullptr, tokens, tokens, 512, 512, 256, 256);

        if (i % 3 == 0) {
            gemm_wmma<0, true, false><<<dim3(DM / 64, NTOK / 128), 256, 0, stream>>>(
                tokens, F(B0 + 23), F(B0 + 22), nullptr, qb, 256, 256, 256, 256);
            gemm_wmma<0, true, false><<<dim3(DM / 64, NTOK / 128), 256, 0, stream>>>(
                sar_tok, F(B0 + 17), F(B0 + 16), nullptr, kb, 256, 256, 256, 256);
            gemm_wmma<0, true, false><<<dim3(DM / 64, NTOK / 128), 256, 0, stream>>>(
                sar_tok, F(B0 + 25), F(B0 + 24), nullptr, vb, 256, 256, 256, 256);
            attn_wmma<<<B_SZ * NH * 8, 256, 0, stream>>>(qb, kb, vb, ab);
            gemm_wmma<0, true, true><<<dim3(DM / 64, NTOK / 128), 256, 0, stream>>>(
                ab, F(B0 + 21), F(B0 + 20), tokens, tokens, 256, 256, 256, 256);
            ln_kernel<<<512, 256, 0, stream>>>(tokens, F(B0 + 19), F(B0 + 18), tokens);
        }
    }

    ln_kernel<<<512, 256, 0, stream>>>(tokens, F(237), F(236), tokens);
    unembed_kernel<<<(NTOK * 13 + 255) / 256, 256, 0, stream>>>(
        tokens, F(241), F(240), (float*)d_out);
}
